// TopoRobustGNN_77352361001296
// MI455X (gfx1250) — matrix operand order repaired and checked by reading, not compiled, not run on hardware
//
#include <hip/hip_runtime.h>
#include <hip/hip_bf16.h>
#include <string.h>

// ---------------------------------------------------------------------------
// TopoRobustGNN forward for MI455X (gfx1250, wave32, WMMA).
// Hot loop: regularizer edge-MLP (E=262144 x [512->256]) ~69 GFLOP done with
// v_wmma_f32_16x16x32_f16; activations carried as f16 (halves bytes moved,
// 23.3 TB/s HBM makes the edge gather the bound once WMMA is used).  Edge
// pair tiles are staged into LDS with global_load_async_to_lds_b128.
// ---------------------------------------------------------------------------

#define NN   8192
#define EE   262144
#define ET   (EE + NN)      // edges + self loops
#define FIN  128
#define HH   4
#define CC   64
#define HC   256
#define EDD  3
#define NC   8
#define RN   8192           // adjacency dim

typedef __attribute__((ext_vector_type(16))) _Float16 v16h;
typedef __attribute__((ext_vector_type(8)))  _Float16 v8h;
typedef __attribute__((ext_vector_type(8)))  float    v8f;

__device__ __forceinline__ unsigned fenc(float f) {
  unsigned u = __float_as_uint(f);
  return (u >> 31) ? ~u : (u | 0x80000000u);
}
__device__ __forceinline__ float fdec(unsigned u) {
  return (u & 0x80000000u) ? __uint_as_float(u & 0x7FFFFFFFu) : __uint_as_float(~u);
}
__device__ __forceinline__ float eluf(float x) { return x > 0.f ? x : (__expf(x) - 1.f); }
__device__ __forceinline__ float sigmf(float x) { return 1.f / (1.f + __expf(-x)); }

// ---------------------------------------------------------------------------
// Vectorized zero (n must be a multiple of 4 floats)
// ---------------------------------------------------------------------------
__global__ void k_zero4(float4* p, size_t n4) {
  size_t i = (size_t)blockIdx.x * blockDim.x + threadIdx.x;
  size_t st = (size_t)gridDim.x * blockDim.x;
  float4 z; z.x = 0.f; z.y = 0.f; z.z = 0.f; z.w = 0.f;
  for (; i < n4; i += st) p[i] = z;
}

// ---------------------------------------------------------------------------
// Generic WMMA GEMM: D[M,Ncols] = act(A[M,K] @ B[K,Ncols] + bias)
// One wave32 per 16x16 output tile; A rows loaded as float4 (b128) and packed
// to f16; B columns loaded unconditionally (clamped index + 0/1 mask) so the
// inner loop is branch-free.  K % 32 == 0, M % 16 == 0.
// ---------------------------------------------------------------------------
__global__ void k_gemm(const float* __restrict__ A, const float* __restrict__ B,
                       const float* __restrict__ bias, float* __restrict__ D,
                       int M, int K, int Ncols, int ldd, int act) {
  const int lane = threadIdx.x;
  const int l15  = lane & 15;
  const int hiA  = (lane >= 16) ? 8 : 0;   // A frag: lanes 16-31 carry K+8..15, K+24..31
  const int hiB  = (lane >= 16) ? 16 : 0;  // unused; B frag indexes with hiA, same k as A
  const int m    = blockIdx.y * 16 + l15;
  const int n    = blockIdx.x * 16 + l15;
  const int nc   = (n < Ncols) ? n : (Ncols - 1);
  const float bm = (n < Ncols) ? 1.f : 0.f;
  const float* arow = A + (size_t)m * K;
  v8f c = {};
  for (int kb = 0; kb < K; kb += 32) {
    const float* ap = arow + kb + hiA;
    float4 q0 = *(const float4*)(ap);
    float4 q1 = *(const float4*)(ap + 4);
    float4 q2 = *(const float4*)(ap + 16);
    float4 q3 = *(const float4*)(ap + 20);
    v16h a;
    a[0]  = (_Float16)q0.x; a[1]  = (_Float16)q0.y;
    a[2]  = (_Float16)q0.z; a[3]  = (_Float16)q0.w;
    a[4]  = (_Float16)q1.x; a[5]  = (_Float16)q1.y;
    a[6]  = (_Float16)q1.z; a[7]  = (_Float16)q1.w;
    a[8]  = (_Float16)q2.x; a[9]  = (_Float16)q2.y;
    a[10] = (_Float16)q2.z; a[11] = (_Float16)q2.w;
    a[12] = (_Float16)q3.x; a[13] = (_Float16)q3.y;
    a[14] = (_Float16)q3.z; a[15] = (_Float16)q3.w;
    v16h b;
#pragma unroll
    for (int e = 0; e < 16; ++e) {
      b[e] = (_Float16)(B[(size_t)(kb + hiA + e + ((e >> 3) << 3)) * Ncols + nc] * bm);
    }
    c = __builtin_amdgcn_wmma_f32_16x16x32_f16(false, a, false, b, (short)0, c,
                                               false, false);
  }
  if (n < Ncols) {
    const float bv = bias ? bias[n] : 0.0f;   // hoisted: one load, no branches
#pragma unroll
    for (int v = 0; v < 8; ++v) {
      int mm = blockIdx.y * 16 + v + hiA;
      float val = c[v] + bv;
      if (act == 1) val = eluf(val);
      D[(size_t)mm * ldd + n] = val;
    }
  }
}

// ---------------------------------------------------------------------------
// GAT pieces
// ---------------------------------------------------------------------------
__global__ void k_asrcdst(const float* __restrict__ xs, const float* __restrict__ as,
                          const float* __restrict__ ad, float* osrc, float* odst) {
  int i = blockIdx.x * blockDim.x + threadIdx.x;
  if (i >= NN * HH) return;
  int n = i >> 2, h = i & 3;
  const float* xr = xs + (size_t)n * HC + h * CC;
  float s = 0.f, d = 0.f;
  for (int c = 0; c < CC; ++c) { s += xr[c] * as[h * CC + c]; d += xr[c] * ad[h * CC + c]; }
  osrc[i] = s; odst[i] = d;
}

__global__ void k_eamean(const float* __restrict__ ea, float* out) {
  __shared__ float s[3][256];
  int t = threadIdx.x;
  float a0 = 0.f, a1 = 0.f, a2 = 0.f;
  for (int e = t; e < EE; e += 256) {
    a0 += ea[(size_t)e * 3];
    a1 += ea[(size_t)e * 3 + 1];
    a2 += ea[(size_t)e * 3 + 2];
  }
  s[0][t] = a0; s[1][t] = a1; s[2][t] = a2;
  __syncthreads();
  for (int st = 128; st > 0; st >>= 1) {
    if (t < st) { s[0][t] += s[0][t + st]; s[1][t] += s[1][t + st]; s[2][t] += s[2][t + st]; }
    __syncthreads();
  }
  if (t < 3) out[t] = s[t][0] / (float)EE;
}

__global__ void k_ae(const float* __restrict__ ea, const float* __restrict__ eam,
                     const float* __restrict__ We, const float* __restrict__ atte,
                     float* __restrict__ ae_out) {
  int i = blockIdx.x * blockDim.x + threadIdx.x;
  if (i >= ET * HH) return;
  int e = i >> 2, h = i & 3;
  float e0, e1, e2;
  if (e < EE) { e0 = ea[(size_t)e * 3]; e1 = ea[(size_t)e * 3 + 1]; e2 = ea[(size_t)e * 3 + 2]; }
  else        { e0 = eam[0]; e1 = eam[1]; e2 = eam[2]; }
  float acc = 0.f;
  for (int c = 0; c < CC; ++c) {
    int col = h * CC + c;
    float xe = e0 * We[col] + e1 * We[HC + col] + e2 * We[2 * HC + col];
    acc += xe * atte[h * CC + c];
  }
  ae_out[i] = acc;
}

__global__ void k_alphamax(const float* __restrict__ as, const float* __restrict__ ad,
                           const float* __restrict__ ae, const int* __restrict__ ei,
                           float* __restrict__ alpha, unsigned* __restrict__ amax) {
  int i = blockIdx.x * blockDim.x + threadIdx.x;
  if (i >= ET * HH) return;
  int e = i >> 2, h = i & 3;
  int r = (e < EE) ? ei[e] : (e - EE);
  int c = (e < EE) ? ei[EE + e] : (e - EE);
  float a = as[r * HH + h] + ad[c * HH + h] + ae[i];
  a = a > 0.f ? a : 0.2f * a;              // leaky relu
  alpha[i] = a;
  atomicMax(&amax[c * HH + h], fenc(a));
}

__global__ void k_alphaexp(const int* __restrict__ ei, const unsigned* __restrict__ amax,
                           float* __restrict__ alpha, float* __restrict__ den) {
  int i = blockIdx.x * blockDim.x + threadIdx.x;
  if (i >= ET * HH) return;
  int e = i >> 2, h = i & 3;
  int c = (e < EE) ? ei[EE + e] : (e - EE);
  float v = __expf(alpha[i] - fdec(amax[c * HH + h]));
  alpha[i] = v;
  atomicAdd(&den[c * HH + h], v);
}

__global__ void k_gat_scatter(const float* __restrict__ xs, const float* __restrict__ alpha,
                              const float* __restrict__ den, const int* __restrict__ ei,
                              float* __restrict__ agg) {
  int e = blockIdx.x, t = threadIdx.x;
  int r = (e < EE) ? ei[e] : (e - EE);
  int c = (e < EE) ? ei[EE + e] : (e - EE);
  int h = t >> 6;
  __builtin_prefetch(&xs[(size_t)r * HC + t], 0, 0);
  float al = alpha[e * HH + h] / (den[c * HH + h] + 1e-16f);
  atomicAdd(&agg[(size_t)c * HC + t], xs[(size_t)r * HC + t] * al);
}

// ---------------------------------------------------------------------------
// bias + (optional) ELU + LayerNorm; optional f16 shadow copy for WMMA.
// One block per node, blockDim == F (64 or 256).
// ---------------------------------------------------------------------------
__global__ void k_postln(const float* __restrict__ in, const float* __restrict__ bias,
                         const float* __restrict__ g, const float* __restrict__ b,
                         float* __restrict__ out, _Float16* __restrict__ outh,
                         int F, int doelu) {
  __shared__ float red[256];
  int n = blockIdx.x, t = threadIdx.x;
  float x = in[(size_t)n * F + t] + (bias ? bias[t] : 0.f);
  if (doelu) x = eluf(x);
  red[t] = x;
  __syncthreads();
  for (int s = F >> 1; s > 0; s >>= 1) { if (t < s) red[t] += red[t + s]; __syncthreads(); }
  float mu = red[0] / (float)F;
  __syncthreads();
  float d = x - mu;
  red[t] = d * d;
  __syncthreads();
  for (int s = F >> 1; s > 0; s >>= 1) { if (t < s) red[t] += red[t + s]; __syncthreads(); }
  float y = d * rsqrtf(red[0] / (float)F + 1e-5f) * g[t] + b[t];
  out[(size_t)n * F + t] = y;
  if (outh) outh[(size_t)n * F + t] = (_Float16)y;
}

__global__ void k_f16cvt(const float* __restrict__ in, _Float16* __restrict__ out, int n) {
  int i = blockIdx.x * blockDim.x + threadIdx.x;
  if (i < n) out[i] = (_Float16)in[i];
}

// ---------------------------------------------------------------------------
// Regularizer: per 16-edge tile, pair = [h[row]||h[col]] (K=512, f16) staged
// into LDS with global_load_async_to_lds_b128 (ASYNCcnt-tracked, no VGPR
// round trip); 8 wave32s each own 2 column tiles of the 512x256 W1 -> 16
// WMMA 16x16x32 ops per k-step, A frags read as 2x ds b128; then per-edge
// ELU + dot(W2) + sigmoid -> adj.
// ---------------------------------------------------------------------------
__global__ void k_reg(const _Float16* __restrict__ h16, const _Float16* __restrict__ w1,
                      const float* __restrict__ b1, const float* __restrict__ w2,
                      const float* __restrict__ b2, const int* __restrict__ ei,
                      float* __restrict__ adj) {
  __shared__ _Float16 sA[16 * 512];   // 16 KB
  __shared__ float    sC[16 * 256];   // 16 KB
  __shared__ int sR[16], sCl[16];
  const int tid = threadIdx.x;
  const int e0 = blockIdx.x * 16;
  if (tid < 16) { sR[tid] = ei[e0 + tid]; sCl[tid] = ei[EE + e0 + tid]; }
  __syncthreads();
  // 16 edges x 512 halves = 1024 uint4 chunks; row/col halves (256 each) are
  // uint4-aligned, so each 16-byte chunk comes from a single node row.
  // Async DMA each chunk memory -> LDS (lane-wise b128, tracked by ASYNCcnt).
  {
    const uint4* src = (const uint4*)h16;          // node row = 32 uint4
    uint4* dst = (uint4*)sA;
#pragma unroll 4
    for (int idx = tid; idx < 1024; idx += 256) {
      int el = idx >> 6;                 // edge slot (64 uint4 per pair-row)
      int q  = idx & 63;                 // uint4 within pair-row
      int node = (q < 32) ? sR[el] : sCl[el];
      const uint4* gp = src + (size_t)node * 32 + (q & 31);
      unsigned lds = (unsigned)(unsigned long long)(&dst[idx]);  // LDS byte offset
      asm volatile("global_load_async_to_lds_b128 %0, %1, off"
                   :: "v"(lds), "v"((unsigned long long)gp)
                   : "memory");
    }
  }
  asm volatile("s_wait_asynccnt 0x0" ::: "memory");
  __syncthreads();
  const int wave = tid >> 5, lane = tid & 31;
  const int l15 = lane & 15;
  const int hiA = (lane >= 16) ? 8 : 0;
  const int hiB = (lane >= 16) ? 16 : 0;
  const int ct0 = wave * 2, ct1 = wave * 2 + 1;
  v8f c0 = {}, c1 = {};
  for (int kb = 0; kb < 512; kb += 32) {
    const _Float16* arow = sA + (l15 << 9) + kb + hiA;
    v8h lo = *(const v8h*)(arow);        // K+0..7   -> ds_load_b128
    v8h hi = *(const v8h*)(arow + 16);   // K+16..23 -> ds_load_b128
    v16h a = __builtin_shufflevector(lo, hi, 0, 1, 2, 3, 4, 5, 6, 7,
                                     8, 9, 10, 11, 12, 13, 14, 15);
    v16h bb0, bb1;
#pragma unroll
    for (int e = 0; e < 16; ++e) {
      int kk = kb + hiA + e + ((e >> 3) << 3);
      bb0[e] = w1[(size_t)kk * HC + ct0 * 16 + l15];
      bb1[e] = w1[(size_t)kk * HC + ct1 * 16 + l15];
    }
    c0 = __builtin_amdgcn_wmma_f32_16x16x32_f16(false, a, false, bb0, (short)0, c0, false, false);
    c1 = __builtin_amdgcn_wmma_f32_16x16x32_f16(false, a, false, bb1, (short)0, c1, false, false);
  }
#pragma unroll
  for (int v = 0; v < 8; ++v) {
    int mm = v + hiA;
    sC[mm * 256 + ct0 * 16 + l15] = c0[v];
    sC[mm * 256 + ct1 * 16 + l15] = c1[v];
  }
  __syncthreads();
  if (tid < 16) {
    float acc = b2[0];
    for (int cc = 0; cc < HC; ++cc) {
      float hv = eluf(sC[tid * 256 + cc] + b1[cc]);
      acc += hv * w2[cc];
    }
    adj[(size_t)sR[tid] * RN + sCl[tid]] = sigmf(acc);
  }
}

__global__ void k_diag(float* __restrict__ adj) {
  int i = blockIdx.x * blockDim.x + threadIdx.x;
  if (i < NN) adj[(size_t)i * (RN + 1)] = 1.0f;
}

// ---------------------------------------------------------------------------
// Confidence GCN pieces
// ---------------------------------------------------------------------------
__global__ void k_deg(const int* __restrict__ ei, float* __restrict__ deg) {
  int e = blockIdx.x * blockDim.x + threadIdx.x;
  if (e < EE) atomicAdd(&deg[ei[e]], 1.0f);
}

__global__ void k_dinv(const float* __restrict__ deg, float* __restrict__ dinv) {
  int i = blockIdx.x * blockDim.x + threadIdx.x;
  if (i < NN) {
    float d = deg[i];
    dinv[i] = d > 0.f ? rsqrtf(fmaxf(d, 1e-12f)) : 0.f;
  }
}

__global__ void k_conf(const float* __restrict__ ea, const float* __restrict__ cw,
                       const float* __restrict__ cb, const float* __restrict__ dinv,
                       const float* __restrict__ adj, const int* __restrict__ ei,
                       float* __restrict__ wn) {
  int e = blockIdx.x * blockDim.x + threadIdx.x;
  if (e >= EE) return;
  int r = ei[e], c = ei[EE + e];
  float s = sigmf(ea[(size_t)e * 3] * cw[0] + ea[(size_t)e * 3 + 1] * cw[1] +
                  ea[(size_t)e * 3 + 2] * cw[2] + cb[0]);
  float w = (s + adj[(size_t)r * RN + c]) * 0.5f;
  wn[e] = dinv[r] * dinv[c] * w;
}

__global__ void k_gcn_scatter(const float* __restrict__ h, const float* __restrict__ wn,
                              const int* __restrict__ ei, float* __restrict__ agg) {
  int e = blockIdx.x, t = threadIdx.x, F = blockDim.x;
  int r = ei[e], c = ei[EE + e];
  __builtin_prefetch(&h[(size_t)r * F + t], 0, 0);
  atomicAdd(&agg[(size_t)c * F + t], h[(size_t)r * F + t] * wn[e]);
}

// ---------------------------------------------------------------------------
// Host-side launch sequence
// ---------------------------------------------------------------------------
extern "C" void kernel_launch(void* const* d_in, const int* in_sizes, int n_in,
                              void* d_out, int out_size, void* d_ws, size_t ws_size,
                              hipStream_t stream) {
  const float* x     = (const float*)d_in[0];
  const float* eattr = (const float*)d_in[1];
  const float* glw   = (const float*)d_in[2];
  const float* glew  = (const float*)d_in[3];
  const float* asr   = (const float*)d_in[4];
  const float* ads   = (const float*)d_in[5];
  const float* aed   = (const float*)d_in[6];
  const float* gbias = (const float*)d_in[7];
  const float* n1g   = (const float*)d_in[8];
  const float* n1b   = (const float*)d_in[9];
  const float* rw1   = (const float*)d_in[10];
  const float* rb1   = (const float*)d_in[11];
  const float* rw2   = (const float*)d_in[12];
  const float* rb2   = (const float*)d_in[13];
  const float* g1w   = (const float*)d_in[14];
  const float* g1b   = (const float*)d_in[15];
  const float* g1cw  = (const float*)d_in[16];
  const float* g1cb  = (const float*)d_in[17];
  const float* n2g   = (const float*)d_in[18];
  const float* n2b   = (const float*)d_in[19];
  const float* g2w   = (const float*)d_in[20];
  const float* g2b   = (const float*)d_in[21];
  const float* g2cw  = (const float*)d_in[22];
  const float* g2cb  = (const float*)d_in[23];
  const float* cw1   = (const float*)d_in[24];
  const float* cb1   = (const float*)d_in[25];
  const float* cw2   = (const float*)d_in[26];
  const float* cb2   = (const float*)d_in[27];
  const int*   ei    = (const int*)d_in[28];

  float* out = (float*)d_out;
  float* adj = out + (size_t)NN * NC;   // logits first, then adjacency
  float* ws  = (float*)d_ws;

  // workspace layout (float offsets)
  const size_t XS   = 0;          // [N,256] GAT projection
  const size_t AGG  = 2097152;    // [N,256] aggregation buffer (reused)
  const size_t H1   = 4194304;    // [N,256] post-LN fp32
  const size_t H1H  = 6291456;    // [N,256] f16 (1,048,576 float slots)
  const size_t ASRC = 7340032;    // [N,H]
  const size_t ADST = 7372800;
  const size_t AE   = 7405568;    // [ET,H]
  const size_t ALPH = 8486912;    // [ET,H]
  const size_t AMAX = 9568256;    // [N,H] ordered-uint
  const size_t DEN  = 9601024;    // [N,H]
  const size_t EAM  = 9633792;    // 8
  const size_t DEG  = 9633800;    // [N]
  const size_t DINV = 9641992;    // [N]
  const size_t W1H  = 9650184;    // 512x256 f16 (65,536 float slots)
  const size_t WN   = 9715720;    // [E]
  const size_t H2   = 9977864;    // [N,64]
  const size_t H3   = 10502152;   // [N,64]
  const size_t TMP  = 11026440;   // [N,64]

  // --- clear accumulators + output -----------------------------------------
  k_zero4<<<64, 256, 0, stream>>>((float4*)(ws + AMAX),
                                  (32768 + 32768 + 8 + 8192) / 4);   // AMAX..DEG
  k_zero4<<<512, 256, 0, stream>>>((float4*)(ws + AGG), ((size_t)NN * HC) / 4);
  k_zero4<<<8192, 256, 0, stream>>>((float4*)out,
                                    ((size_t)NN * NC + (size_t)RN * RN) / 4);

  // --- GAT -----------------------------------------------------------------
  k_gemm<<<dim3(HC / 16, NN / 16), 32, 0, stream>>>(x, glw, nullptr, ws + XS,
                                                    NN, FIN, HC, HC, 0);
  k_asrcdst<<<(NN * HH) / 256, 256, 0, stream>>>(ws + XS, asr, ads, ws + ASRC, ws + ADST);
  k_eamean<<<1, 256, 0, stream>>>(eattr, ws + EAM);
  k_ae<<<(ET * HH) / 256, 256, 0, stream>>>(eattr, ws + EAM, glew, aed, ws + AE);
  k_alphamax<<<(ET * HH) / 256, 256, 0, stream>>>(ws + ASRC, ws + ADST, ws + AE, ei,
                                                  ws + ALPH, (unsigned*)(ws + AMAX));
  k_alphaexp<<<(ET * HH) / 256, 256, 0, stream>>>(ei, (const unsigned*)(ws + AMAX),
                                                  ws + ALPH, ws + DEN);
  k_gat_scatter<<<ET, HC, 0, stream>>>(ws + XS, ws + ALPH, ws + DEN, ei, ws + AGG);
  k_postln<<<NN, HC, 0, stream>>>(ws + AGG, gbias, n1g, n1b, ws + H1,
                                  (_Float16*)(ws + H1H), HC, 1);

  // --- regularizer (dense adjacency) ---------------------------------------
  k_f16cvt<<<512, 256, 0, stream>>>(rw1, (_Float16*)(ws + W1H), 512 * HC);
  k_reg<<<EE / 16, 256, 0, stream>>>((const _Float16*)(ws + H1H),
                                     (const _Float16*)(ws + W1H),
                                     rb1, rw2, rb2, ei, adj);
  k_diag<<<NN / 256, 256, 0, stream>>>(adj);

  // --- degree normalization -------------------------------------------------
  k_deg<<<EE / 256, 256, 0, stream>>>(ei, ws + DEG);
  k_dinv<<<NN / 256, 256, 0, stream>>>(ws + DEG, ws + DINV);

  // --- confidence GCN 1 -----------------------------------------------------
  k_conf<<<EE / 256, 256, 0, stream>>>(eattr, g1cw, g1cb, ws + DINV, adj, ei, ws + WN);
  k_zero4<<<512, 256, 0, stream>>>((float4*)(ws + AGG), ((size_t)NN * HC) / 4);
  k_gcn_scatter<<<EE, HC, 0, stream>>>(ws + H1, ws + WN, ei, ws + AGG);
  k_gemm<<<dim3(CC / 16, NN / 16), 32, 0, stream>>>(ws + AGG, g1w, g1b, ws + TMP,
                                                    NN, HC, CC, CC, 0);
  k_postln<<<NN, CC, 0, stream>>>(ws + TMP, nullptr, n2g, n2b, ws + H2, nullptr, CC, 1);

  // --- confidence GCN 2 -----------------------------------------------------
  k_conf<<<EE / 256, 256, 0, stream>>>(eattr, g2cw, g2cb, ws + DINV, adj, ei, ws + WN);
  k_zero4<<<512, 256, 0, stream>>>((float4*)(ws + AGG), ((size_t)NN * CC) / 4);
  k_gcn_scatter<<<EE, CC, 0, stream>>>(ws + H2, ws + WN, ei, ws + AGG);
  k_gemm<<<dim3(CC / 16, NN / 16), 32, 0, stream>>>(ws + AGG, g2w, g2b, ws + H3,
                                                    NN, CC, CC, CC, 0);

  // --- classifier -> logits into d_out --------------------------------------
  k_gemm<<<dim3(CC / 16, NN / 16), 32, 0, stream>>>(ws + H3, cw1, cb1, ws + TMP,
                                                    NN, CC, CC, CC, 1);
  k_gemm<<<dim3(1, NN / 16), 32, 0, stream>>>(ws + TMP, cw2, cb2, out,
                                              NN, CC, NC, NC, 0);
}
